// MambaDenoiser_1864015806533
// MI455X (gfx1250) — compile-verified, matrix operand order checked by reading, not run on hardware
//
#include <hip/hip_runtime.h>
#include <math.h>

// ---------------- types ----------------
typedef _Float16 v8h  __attribute__((ext_vector_type(8)));
typedef _Float16 v16h __attribute__((ext_vector_type(16)));
typedef float    v8f  __attribute__((ext_vector_type(8)));

#define D_MODEL   1024
#define D_INNER   2048
#define D_STATE   16
#define DT_RANK   64
#define N_LAYERS  8
#define BATCH     16
#define LSEQ      256            // 16x16 tokens
#define ROWS      (BATCH*LSEQ)   // 4096

// ---------------- WMMA GEMM: C[M,N] = A[M,K] * B[N,K]^T ----------------
// Double-buffered LDS, one barrier per K-step; global loads for tile k+1
// issued before the WMMA block of tile k so they hide under matrix math.
// MODE 0: none, 1: bias+softplus, 2: bias+relu
template<int MODE>
__global__ __launch_bounds__(256)
void wmma_gemm_kernel(const _Float16* __restrict__ A,
                      const _Float16* __restrict__ B,
                      const float*    __restrict__ bias,
                      float*          __restrict__ C,
                      int M, int N, int K)
{
    constexpr int BM = 128, BN = 128, BK = 32;
    __shared__ _Float16 As[2][BM][BK + 8];
    __shared__ _Float16 Bs[2][BN][BK + 8];

    const int tid  = threadIdx.x;
    const int lane = tid & 31;
    const int wave = tid >> 5;
    const int wm   = (wave >> 2) * 64;   // 0 or 64
    const int wn   = (wave & 3) * 32;    // 0,32,64,96
    const int m0   = blockIdx.y * BM;
    const int n0   = blockIdx.x * BN;

    const int lrow = lane & 15;
    const int lhi  = lane >> 4;
    const int kA   = lhi * 8;            // halves 0..7 at kA, 8..15 at 16+kA

    // cooperative-load coordinates: thread covers rows r0 and r0+64 at col kc
    const int r0 = tid >> 2;             // 0..63
    const int kc = (tid & 3) * 8;

    v8f acc[4][2] = {};

    const int nk = K / BK;

    // ---- preload tile 0 into buffer 0 ----
    {
        *(uint4*)&As[0][r0][kc]      = *(const uint4*)(A + (size_t)(m0 + r0)      * K + kc);
        *(uint4*)&As[0][r0 + 64][kc] = *(const uint4*)(A + (size_t)(m0 + r0 + 64) * K + kc);
        uint4 b0 = {0u,0u,0u,0u}, b1 = {0u,0u,0u,0u};
        if (n0 + r0      < N) b0 = *(const uint4*)(B + (size_t)(n0 + r0)      * K + kc);
        if (n0 + r0 + 64 < N) b1 = *(const uint4*)(B + (size_t)(n0 + r0 + 64) * K + kc);
        *(uint4*)&Bs[0][r0][kc]      = b0;
        *(uint4*)&Bs[0][r0 + 64][kc] = b1;
    }
    __syncthreads();

    for (int k = 0; k < nk; ++k) {
        const int cur = k & 1;
        const int nxt = cur ^ 1;
        const bool havenext = (k + 1) < nk;

        // issue global loads for tile k+1 (latency hidden by WMMAs below)
        uint4 a0v, a1v, b0v, b1v;
        if (havenext) {
            const int kb = (k + 1) * BK;
            a0v = *(const uint4*)(A + (size_t)(m0 + r0)      * K + kb + kc);
            a1v = *(const uint4*)(A + (size_t)(m0 + r0 + 64) * K + kb + kc);
            b0v = (uint4){0u,0u,0u,0u};
            b1v = (uint4){0u,0u,0u,0u};
            if (n0 + r0      < N) b0v = *(const uint4*)(B + (size_t)(n0 + r0)      * K + kb + kc);
            if (n0 + r0 + 64 < N) b1v = *(const uint4*)(B + (size_t)(n0 + r0 + 64) * K + kb + kc);
        }

        // ---- compute on buffer `cur` ----
        v16h bfrag[2];
        #pragma unroll
        for (int ni = 0; ni < 2; ++ni) {
            int n = wn + ni * 16 + lrow;
            v8h lo = *(const v8h*)&Bs[cur][n][kA];
            v8h hi = *(const v8h*)&Bs[cur][n][16 + kA];
            #pragma unroll
            for (int j = 0; j < 8; ++j) { bfrag[ni][j] = lo[j]; bfrag[ni][8 + j] = hi[j]; }
        }
        #pragma unroll
        for (int mi = 0; mi < 4; ++mi) {
            int m = wm + mi * 16 + lrow;
            v8h lo = *(const v8h*)&As[cur][m][kA];
            v8h hi = *(const v8h*)&As[cur][m][16 + kA];
            v16h afrag;
            #pragma unroll
            for (int j = 0; j < 8; ++j) { afrag[j] = lo[j]; afrag[8 + j] = hi[j]; }
            #pragma unroll
            for (int ni = 0; ni < 2; ++ni) {
                acc[mi][ni] = __builtin_amdgcn_wmma_f32_16x16x32_f16(
                    false, afrag, false, bfrag[ni], (short)0, acc[mi][ni], false, false);
            }
        }

        // ---- stage tile k+1 into the other buffer ----
        if (havenext) {
            *(uint4*)&As[nxt][r0][kc]      = a0v;
            *(uint4*)&As[nxt][r0 + 64][kc] = a1v;
            *(uint4*)&Bs[nxt][r0][kc]      = b0v;
            *(uint4*)&Bs[nxt][r0 + 64][kc] = b1v;
        }
        __syncthreads();
    }

    // epilogue + store (C/D layout: lane (lrow,lhi), vgpr r -> M = lhi*8+r, N = lrow)
    #pragma unroll
    for (int mi = 0; mi < 4; ++mi) {
        #pragma unroll
        for (int ni = 0; ni < 2; ++ni) {
            int ncol = n0 + wn + ni * 16 + lrow;
            if (ncol >= N) continue;
            float bv = (MODE != 0) ? bias[ncol] : 0.f;
            #pragma unroll
            for (int r = 0; r < 8; ++r) {
                int   mrow = m0 + wm + mi * 16 + lhi * 8 + r;
                float v    = acc[mi][ni][r];
                if (MODE == 1) { v += bv; v = (v > 20.f) ? v : log1pf(__expf(v)); }
                else if (MODE == 2) { v += bv; v = v > 0.f ? v : 0.f; }
                C[(size_t)mrow * N + ncol] = v;
            }
        }
    }
}

// ---------------- elementwise f32 -> f16 convert ----------------
__global__ void cvt_f16_kernel(const float* __restrict__ in, _Float16* __restrict__ out, size_t n)
{
    size_t i = (size_t)blockIdx.x * 256 + threadIdx.x;
    if (i < n) out[i] = (_Float16)in[i];
}

// ---------------- patch embed + pixel_unshuffle -> sequence ----------------
__global__ __launch_bounds__(256)
void pe_kernel(const float* __restrict__ x, const float* __restrict__ noise,
               const float* __restrict__ w1, const float* __restrict__ b1,
               const float* __restrict__ w2, float* __restrict__ seq0)
{
    int pix = blockIdx.x;               // b*1024 + h*32 + w
    int b = pix >> 10;
    int hw = pix & 1023;
    int h = hw >> 5, w = hw & 31;
    __shared__ float t[256];
    int o = threadIdx.x;
    float in0 = x[((b * 3 + 0) * 32 + h) * 32 + w];
    float in1 = x[((b * 3 + 1) * 32 + h) * 32 + w];
    float in2 = x[((b * 3 + 2) * 32 + h) * 32 + w];
    float in3 = noise[b];
    float s = w1[o * 4 + 0] * in0 + w1[o * 4 + 1] * in1 +
              w1[o * 4 + 2] * in2 + w1[o * 4 + 3] * in3 + b1[o];
    t[o] = s > 0.f ? s : 0.f;
    __syncthreads();
    float acc = 0.f;
    const float* w2r = w2 + o * 256;
    for (int c = 0; c < 256; ++c) acc += w2r[c] * t[c];
    int l  = (h >> 1) * 16 + (w >> 1);
    int ch = o * 4 + (h & 1) * 2 + (w & 1);
    seq0[((size_t)(b * 256 + l)) * D_MODEL + ch] = acc;
}

// ---------------- residual add + layernorm -> f16 ----------------
__global__ __launch_bounds__(256)
void addln_kernel(const float* __restrict__ y, float* __restrict__ z,
                  const float* __restrict__ lnw, const float* __restrict__ lnb,
                  _Float16* __restrict__ yn16, int first)
{
    int row = blockIdx.x;                  // b*256 + l
    const float* yr = y + (size_t)row * D_MODEL;
    float*       zr = z + (size_t)row * D_MODEL;
    int t = threadIdx.x;
    float v[4];
    float lsum = 0.f, lsq = 0.f;
    #pragma unroll
    for (int i = 0; i < 4; ++i) {
        int c = t + i * 256;
        float val = yr[c] + (first ? 0.f : zr[c]);
        v[i] = val; lsum += val; lsq += val * val;
    }
    __shared__ float s1[256], s2[256];
    s1[t] = lsum; s2[t] = lsq;
    __syncthreads();
    for (int off = 128; off > 0; off >>= 1) {
        if (t < off) { s1[t] += s1[t + off]; s2[t] += s2[t + off]; }
        __syncthreads();
    }
    float mu   = s1[0] * (1.f / D_MODEL);
    float var  = s2[0] * (1.f / D_MODEL) - mu * mu;
    float rstd = rsqrtf(var + 1e-5f);
    #pragma unroll
    for (int i = 0; i < 4; ++i) {
        int c = t + i * 256;
        zr[c] = v[i];
        float nv = (v[i] - mu) * rstd * lnw[c] + lnb[c];
        yn16[(size_t)row * D_MODEL + c] = (_Float16)nv;
    }
}

// ---------------- depthwise causal conv (k=4) + SiLU ----------------
__global__ void conv_silu_kernel(const float* __restrict__ xz,
                                 const float* __restrict__ cw, const float* __restrict__ cb,
                                 float* __restrict__ xact, _Float16* __restrict__ xact16)
{
    size_t idx = (size_t)blockIdx.x * 256 + threadIdx.x;   // over ROWS*D_INNER
    if (idx >= (size_t)ROWS * D_INNER) return;
    int    c    = (int)(idx & (D_INNER - 1));
    size_t bl   = idx >> 11;           // b*256 + l
    int    l    = (int)(bl & 255);
    size_t brow = bl - l;
    float acc = cb[c];
    #pragma unroll
    for (int k = 0; k < 4; ++k) {
        int lt = l - 3 + k;
        if (lt >= 0) acc += cw[c * 4 + k] * xz[(brow + lt) * (2 * D_INNER) + c];
    }
    float sv = acc / (1.f + __expf(-acc));
    xact[idx]   = sv;
    xact16[idx] = (_Float16)sv;
}

// ---------------- extract dbc[:, :64] as f16 ----------------
__global__ void dbc16_kernel(const float* __restrict__ dbc, _Float16* __restrict__ out)
{
    size_t i = (size_t)blockIdx.x * 256 + threadIdx.x;   // ROWS*64
    if (i >= (size_t)ROWS * DT_RANK) return;
    size_t r = i >> 6;
    int    c = (int)(i & 63);
    out[i] = (_Float16)dbc[r * 96 + c];
}

// ---------------- selective scan + gate ----------------
__global__ __launch_bounds__(256)
void scan_kernel(const float* __restrict__ dtbuf, const float* __restrict__ dbc,
                 const float* __restrict__ xact,  const float* __restrict__ xz,
                 const float* __restrict__ A_log, const float* __restrict__ Dskip,
                 _Float16* __restrict__ yact16)
{
    int b  = blockIdx.x >> 3;
    int cg = blockIdx.x & 7;
    int d  = cg * 256 + threadIdx.x;
    float A[D_STATE], h[D_STATE];
    #pragma unroll
    for (int s = 0; s < D_STATE; ++s) { A[s] = -__expf(A_log[d * D_STATE + s]); h[s] = 0.f; }
    float dsk = Dskip[d];
    __shared__ float BC[32];
    size_t base = (size_t)b * 256;
    for (int l = 0; l < 256; ++l) {
        size_t row = base + l;
        if (threadIdx.x < 32) BC[threadIdx.x] = dbc[row * 96 + DT_RANK + threadIdx.x];
        __syncthreads();
        float dt = dtbuf[row * D_INNER + d];
        float xv = xact[row * D_INNER + d];
        float dx = dt * xv;
        float y  = 0.f;
        #pragma unroll
        for (int s = 0; s < D_STATE; ++s) {
            h[s] = __expf(dt * A[s]) * h[s] + dx * BC[s];
            y += h[s] * BC[16 + s];
        }
        float zv  = xz[row * (2 * D_INNER) + D_INNER + d];
        float out = (y + xv * dsk) * (zv / (1.f + __expf(-zv)));
        yact16[row * D_INNER + d] = (_Float16)out;
        __syncthreads();
    }
}

// ---------------- spatial transpose (16x16) + optional sequence reverse ----------------
__global__ __launch_bounds__(256)
void permute_kernel(const float* __restrict__ src, float* __restrict__ dst, int rev)
{
    int l  = blockIdx.x & 255;
    int b  = blockIdx.x >> 8;
    int lt = rev ? (255 - l) : l;
    int sl = (lt & 15) * 16 + (lt >> 4);
    const float4* s  = (const float4*)(src + ((size_t)(b * 256 + sl)) * D_MODEL);
    float4*       dp = (float4*)(dst + ((size_t)(b * 256 + l)) * D_MODEL);
    dp[threadIdx.x] = s[threadIdx.x];
}

// ---------------- build [X ; y ; z] concat as f16 A-matrix ----------------
__global__ void build_ain_kernel(const float* __restrict__ seq0, const float* __restrict__ y,
                                 const float* __restrict__ z, _Float16* __restrict__ ain)
{
    size_t i = (size_t)blockIdx.x * 256 + threadIdx.x;   // ROWS*3072
    if (i >= (size_t)ROWS * 3072) return;
    size_t row = i / 3072;
    int    k   = (int)(i % 3072);
    float v;
    if (k < 1024)      v = seq0[row * D_MODEL + k];
    else if (k < 2048) v = y[row * D_MODEL + (k - 1024)];
    else               v = z[row * D_MODEL + (k - 2048)];
    ain[i] = (_Float16)v;
}

// ---------------- pixel_shuffle + final 1x1 conv ----------------
__global__ __launch_bounds__(256)
void final_kernel(const float* __restrict__ t, const float* __restrict__ w2,
                  const float* __restrict__ b2, float* __restrict__ out)
{
    int idx = blockIdx.x * 256 + threadIdx.x;   // 16*32*32
    if (idx >= BATCH * 32 * 32) return;
    int b = idx >> 10;
    int h = (idx >> 5) & 31;
    int w = idx & 31;
    int l   = (h >> 1) * 16 + (w >> 1);
    int sub = (h & 1) * 2 + (w & 1);
    const float* trow = t + ((size_t)(b * 256 + l)) * 1024 + sub;
    float a0 = b2[0], a1 = b2[1], a2 = b2[2];
    for (int co = 0; co < 256; ++co) {
        float v = trow[co * 4];
        a0 += w2[0 * 256 + co] * v;
        a1 += w2[1 * 256 + co] * v;
        a2 += w2[2 * 256 + co] * v;
    }
    out[((b * 3 + 0) * 32 + h) * 32 + w] = a0;
    out[((b * 3 + 1) * 32 + h) * 32 + w] = a1;
    out[((b * 3 + 2) * 32 + h) * 32 + w] = a2;
}

// ================= host side =================
static inline size_t align_up(size_t v, size_t a) { return (v + a - 1) & ~(a - 1); }

extern "C" void kernel_launch(void* const* d_in, const int* in_sizes, int n_in,
                              void* d_out, int out_size, void* d_ws, size_t ws_size,
                              hipStream_t stream)
{
    (void)in_sizes; (void)n_in; (void)out_size; (void)ws_size;
    const float* x      = (const float*)d_in[0];
    const float* noise  = (const float*)d_in[1];
    const float* pe_w1  = (const float*)d_in[2];
    const float* pe_b1  = (const float*)d_in[3];
    const float* pe_w2  = (const float*)d_in[4];
    const float* ln_w   = (const float*)d_in[5];
    const float* ln_b   = (const float*)d_in[6];
    const float* W_in   = (const float*)d_in[7];
    const float* conv_w = (const float*)d_in[8];
    const float* conv_b = (const float*)d_in[9];
    const float* W_x    = (const float*)d_in[10];
    const float* W_dt   = (const float*)d_in[11];
    const float* b_dt   = (const float*)d_in[12];
    const float* A_log  = (const float*)d_in[13];
    const float* D_skip = (const float*)d_in[14];
    const float* W_out  = (const float*)d_in[15];
    const float* op_w1  = (const float*)d_in[16];
    const float* op_b1  = (const float*)d_in[17];
    const float* op_w2  = (const float*)d_in[18];
    const float* op_b2  = (const float*)d_in[19];
    float* out = (float*)d_out;

    // ---- workspace carve-up ----
    char*  base = (char*)d_ws;
    size_t off  = 0;
    auto alloc = [&](size_t bytes) { void* p = base + off; off = align_up(off + bytes, 256); return p; };

    const size_t nWin  = (size_t)N_LAYERS * 2 * D_INNER * D_MODEL;   // 8*4096*1024
    const size_t nWx   = (size_t)N_LAYERS * 96 * D_INNER;
    const size_t nWdt  = (size_t)N_LAYERS * D_INNER * DT_RANK;
    const size_t nWout = (size_t)N_LAYERS * D_MODEL * D_INNER;
    const size_t nOp1  = (size_t)1024 * 3072;

    _Float16* Win16  = (_Float16*)alloc(nWin  * 2);
    _Float16* Wx16   = (_Float16*)alloc(nWx   * 2);
    _Float16* Wdt16  = (_Float16*)alloc(nWdt  * 2);
    _Float16* Wout16 = (_Float16*)alloc(nWout * 2);
    _Float16* Op116  = (_Float16*)alloc(nOp1  * 2);

    float*    seq0   = (float*)alloc((size_t)ROWS * D_MODEL * 4);
    float*    yA     = (float*)alloc((size_t)ROWS * D_MODEL * 4);
    float*    yB     = (float*)alloc((size_t)ROWS * D_MODEL * 4);
    float*    zA     = (float*)alloc((size_t)ROWS * D_MODEL * 4);
    float*    zB     = (float*)alloc((size_t)ROWS * D_MODEL * 4);
    _Float16* yn16   = (_Float16*)alloc((size_t)ROWS * D_MODEL * 2);
    float*    xz     = (float*)alloc((size_t)ROWS * 2 * D_INNER * 4);
    float*    xact   = (float*)alloc((size_t)ROWS * D_INNER * 4);
    _Float16* xact16 = (_Float16*)alloc((size_t)ROWS * D_INNER * 2);
    float*    dbc    = (float*)alloc((size_t)ROWS * 96 * 4);
    _Float16* dbc16  = (_Float16*)alloc((size_t)ROWS * DT_RANK * 2);
    float*    dtbuf  = (float*)alloc((size_t)ROWS * D_INNER * 4);
    _Float16* yact16 = (_Float16*)alloc((size_t)ROWS * D_INNER * 2);
    float*    ytmp   = (float*)alloc((size_t)ROWS * D_MODEL * 4);
    _Float16* ain16  = (_Float16*)alloc((size_t)ROWS * 3072 * 2);
    float*    tbuf   = (float*)alloc((size_t)ROWS * 1024 * 4);

    auto cvt = [&](const float* src, _Float16* dst, size_t n) {
        cvt_f16_kernel<<<(unsigned)((n + 255) / 256), 256, 0, stream>>>(src, dst, n);
    };

    // ---- one-time (per call) weight conversion to f16 ----
    cvt(W_in,  Win16,  nWin);
    cvt(W_x,   Wx16,   nWx);
    cvt(W_dt,  Wdt16,  nWdt);
    cvt(W_out, Wout16, nWout);
    cvt(op_w1, Op116,  nOp1);

    // ---- patch embed -> sequence ----
    pe_kernel<<<BATCH * 32 * 32, 256, 0, stream>>>(x, noise, pe_w1, pe_b1, pe_w2, seq0);

    float* ycur = seq0;
    float* zcur = zA;
    float* ybufs[2] = { yA, yB };

    for (int i = 0; i < N_LAYERS; ++i) {
        // residual + layernorm
        addln_kernel<<<ROWS, 256, 0, stream>>>(ycur, zcur,
                                               ln_w + (size_t)i * D_MODEL,
                                               ln_b + (size_t)i * D_MODEL,
                                               yn16, i == 0 ? 1 : 0);
        // xz = yn * W_in^T   (4096 x 4096, K=1024)
        wmma_gemm_kernel<0><<<dim3(32, 32), 256, 0, stream>>>(
            yn16, Win16 + (size_t)i * 2 * D_INNER * D_MODEL, nullptr, xz,
            ROWS, 2 * D_INNER, D_MODEL);
        // depthwise causal conv + SiLU
        conv_silu_kernel<<<(unsigned)(((size_t)ROWS * D_INNER + 255) / 256), 256, 0, stream>>>(
            xz, conv_w + (size_t)i * D_INNER * 4, conv_b + (size_t)i * D_INNER, xact, xact16);
        // dbc = xact * W_x^T   (4096 x 96, K=2048)
        wmma_gemm_kernel<0><<<dim3(1, 32), 256, 0, stream>>>(
            xact16, Wx16 + (size_t)i * 96 * D_INNER, nullptr, dbc,
            ROWS, 96, D_INNER);
        // dt = softplus(dbc[:, :64] * W_dt^T + b_dt)   (4096 x 2048, K=64)
        dbc16_kernel<<<(unsigned)(((size_t)ROWS * DT_RANK + 255) / 256), 256, 0, stream>>>(dbc, dbc16);
        wmma_gemm_kernel<1><<<dim3(16, 32), 256, 0, stream>>>(
            dbc16, Wdt16 + (size_t)i * D_INNER * DT_RANK, b_dt + (size_t)i * D_INNER, dtbuf,
            ROWS, D_INNER, DT_RANK);
        // selective scan + gate
        scan_kernel<<<BATCH * 8, 256, 0, stream>>>(
            dtbuf, dbc, xact, xz,
            A_log + (size_t)i * D_INNER * D_STATE, D_skip + (size_t)i * D_INNER, yact16);
        // out = yact * W_out^T   (4096 x 1024, K=2048)
        wmma_gemm_kernel<0><<<dim3(8, 32), 256, 0, stream>>>(
            yact16, Wout16 + (size_t)i * D_MODEL * D_INNER, nullptr, ytmp,
            ROWS, D_MODEL, D_INNER);
        // transpose_xy (+ reverse every 4th layer) on y and z
        int rev = ((i + 1) % 4 == 0) ? 1 : 0;
        float* ynext = ybufs[i & 1];
        float* znext = (zcur == zA) ? zB : zA;
        permute_kernel<<<BATCH * 256, 256, 0, stream>>>(ytmp, ynext, rev);
        permute_kernel<<<BATCH * 256, 256, 0, stream>>>(zcur, znext, rev);
        ycur = ynext;
        zcur = znext;
    }

    // ---- output head: concat [X, y, z] -> 1x1 conv (relu) ----
    build_ain_kernel<<<(unsigned)(((size_t)ROWS * 3072 + 255) / 256), 256, 0, stream>>>(
        seq0, ycur, zcur, ain16);
    wmma_gemm_kernel<2><<<dim3(8, 32), 256, 0, stream>>>(
        ain16, Op116, op_b1, tbuf, ROWS, 1024, 3072);

    // ---- pixel_shuffle + final projection ----
    final_kernel<<<(BATCH * 32 * 32 + 255) / 256, 256, 0, stream>>>(tbuf, op_w2, op_b2, out);
}
